// GCN_86466281603778
// MI455X (gfx1250) — matrix operand order repaired and checked by reading, not compiled, not run on hardware
//
#include <hip/hip_runtime.h>
#include <hip/hip_bf16.h>
#include <math.h>
#include <float.h>

// ---------------------------------------------------------------------------
// Types for CDNA5 WMMA (gfx1250, wave32)
// ---------------------------------------------------------------------------
typedef __attribute__((ext_vector_type(16))) __bf16 v16bf;
typedef __attribute__((ext_vector_type(8)))  __bf16 v8bf;
typedef __attribute__((ext_vector_type(8)))  float  v8f;

static __device__ __forceinline__ __bf16 f2bf(float f) {
  // native fptrunc: RNE; backend can select v_cvt_pk_bf16_f32 on gfx1250
  return (__bf16)f;
}

static __device__ __forceinline__ unsigned mix32(unsigned x) {
  x ^= x >> 16; x *= 0x7FEB352Du;
  x ^= x >> 15; x *= 0x846CA68Bu;
  x ^= x >> 16;
  return x;
}

// ---------------------------------------------------------------------------
// Utility kernels
// ---------------------------------------------------------------------------
__global__ void zero_f32(float* __restrict__ p, int n) {
  int i = blockIdx.x * blockDim.x + threadIdx.x;
  if (i < n) p[i] = 0.0f;
}

__global__ void degree_kernel(const int* __restrict__ dst, float* __restrict__ deg, int E) {
  int i = blockIdx.x * blockDim.x + threadIdx.x;
  if (i < E) unsafeAtomicAdd(&deg[dst[i]], 1.0f);
}

__global__ void dinv_kernel(float* __restrict__ deg, int Nn) {
  int i = blockIdx.x * blockDim.x + threadIdx.x;
  if (i < Nn) deg[i] = rsqrtf(deg[i] + 1.0f);   // in-place: deg -> d^{-1/2}
}

// ---------------------------------------------------------------------------
// Tiled bf16 WMMA GEMM:  Out[M][N] = A[M][K] @ W[K][N]   (fp32 in/out, bf16 MAC)
// Block: 256 threads = 8 waves. Tile: 64(M) x 64(N), K-step 32, double-buffered
// LDS (one barrier per step). A tile row-major in LDS; B tile TRANSPOSED in
// LDS so every lane's 16-element fragment is two contiguous 16-byte runs
// (ds_load_b128 x2 instead of 16x ds_load_u16).
// ---------------------------------------------------------------------------
#define MBLK 64
#define NBLK 64
#define KSTEP 32
#define LPAD 8   // row padding (elements); row stride = 40 elems = 80 B (16B mult)

__global__ __launch_bounds__(256) void gemm_wmma_bf16(
    const float* __restrict__ A, const float* __restrict__ W,
    float* __restrict__ Out, int M, int K, int N) {
  __shared__ __bf16 As[2][MBLK][KSTEP + LPAD];   // [buf][row][k]
  __shared__ __bf16 Bst[2][NBLK][KSTEP + LPAD];  // [buf][col][k]  (transposed)

  const int t    = threadIdx.x;
  const int lane = t & 31;
  const int w    = t >> 5;      // wave 0..7
  const int tm   = w & 3;       // M tile within block
  const int tg   = w >> 2;      // 0..1  -> N tiles 2*tg, 2*tg+1
  const int g    = lane >> 4;   // lane group
  const int sl   = lane & 15;

  const int m0 = blockIdx.x * MBLK;
  const int n0 = blockIdx.y * NBLK;

  // --- cooperative loader mapping ---
  const int lar = t >> 2;            // A row   0..63
  const int lac = (t & 3) * 8;       // A k-col 0,8,16,24
  const int lbn = t & 63;            // B col   0..63
  const int lbk = (t >> 6) * 8;      // B k     0,8,16,24
  const int grow = m0 + lar;
  const int gcol = n0 + lbn;
  const bool aval = grow < M;
  const bool bval = gcol < N;
  const float* Arow = A + (size_t)grow * K + lac;

  float ar[8], br[8];

  auto prefetch = [&](int k0) {
    if (aval) {
      float4 u = *(const float4*)(Arow + k0);
      float4 v = *(const float4*)(Arow + k0 + 4);
      ar[0] = u.x; ar[1] = u.y; ar[2] = u.z; ar[3] = u.w;
      ar[4] = v.x; ar[5] = v.y; ar[6] = v.z; ar[7] = v.w;
    } else {
#pragma unroll
      for (int j = 0; j < 8; ++j) ar[j] = 0.0f;
    }
    if (bval) {
      const float* wp = W + (size_t)(k0 + lbk) * N + gcol;
#pragma unroll
      for (int j = 0; j < 8; ++j) br[j] = wp[(size_t)j * N];
    } else {
#pragma unroll
      for (int j = 0; j < 8; ++j) br[j] = 0.0f;
    }
  };

  v8f acc0 = {}; v8f acc1 = {};

  prefetch(0);
  int p = 0;
  for (int k0 = 0; k0 < K; k0 += KSTEP) {
    // commit prefetched tile to LDS buffer p (one 16B store each)
    {
      v8bf pa, pb;
#pragma unroll
      for (int j = 0; j < 8; ++j) pa[j] = f2bf(ar[j]);
#pragma unroll
      for (int j = 0; j < 8; ++j) pb[j] = f2bf(br[j]);
      *(v8bf*)&As[p][lar][lac]  = pa;
      *(v8bf*)&Bst[p][lbn][lbk] = pb;
    }
    __syncthreads();

    if (k0 + KSTEP < K) prefetch(k0 + KSTEP);   // overlap next global loads

    // A fragment (16-bit A 16x32, contiguous k order chosen here, same as B):
    // lanes 0-15: elems 0..15 = K 0..15 ; lanes16-31: K 16..31
    const __bf16* arow = &As[p][tm * 16 + sl][0];
    v8bf alo = *(const v8bf*)(arow + 16 * g);
    v8bf ahi = *(const v8bf*)(arow + 16 * g + 8);
    v16bf a = __builtin_shufflevector(alo, ahi,
        0,1,2,3,4,5,6,7,8,9,10,11,12,13,14,15);

    // B fragments (32x16): lanes 0-15 hold K 0..15, lanes 16-31 K 16..31
    const __bf16* b0row = &Bst[p][(2 * tg) * 16 + sl][0];
    const __bf16* b1row = b0row + 16 * (KSTEP + LPAD);
    v8bf t0 = *(const v8bf*)(b0row + 16 * g);
    v8bf t1 = *(const v8bf*)(b0row + 16 * g + 8);
    v16bf bb0 = __builtin_shufflevector(t0, t1,
        0,1,2,3,4,5,6,7,8,9,10,11,12,13,14,15);
    v8bf t2 = *(const v8bf*)(b1row + 16 * g);
    v8bf t3 = *(const v8bf*)(b1row + 16 * g + 8);
    v16bf bb1 = __builtin_shufflevector(t2, t3,
        0,1,2,3,4,5,6,7,8,9,10,11,12,13,14,15);

    acc0 = __builtin_amdgcn_wmma_f32_16x16x32_bf16(false, a, false, bb0,
                                                   (short)0, acc0, false, false);
    acc1 = __builtin_amdgcn_wmma_f32_16x16x32_bf16(false, a, false, bb1,
                                                   (short)0, acc1, false, false);
    p ^= 1;
  }

  // C/D layout: VGPR r -> local row r + 8*g ; col = lane&15
  const int rowBase = m0 + tm * 16 + 8 * g;
  const int col0 = n0 + (2 * tg) * 16 + sl;
  const int col1 = col0 + 16;
#pragma unroll
  for (int r = 0; r < 8; ++r) {
    int row = rowBase + r;
    if (row < M) {
      if (col0 < N) Out[(size_t)row * N + col0] = acc0[r];
      if (col1 < N) Out[(size_t)row * N + col1] = acc1[r];
    }
  }
}

// ---------------------------------------------------------------------------
// Agg[i][c] = Ht[i][c] * dinv[i]^2 + bias[c]   (self-loop term + bias)
// ---------------------------------------------------------------------------
__global__ void init_agg(const float* __restrict__ Ht, const float* __restrict__ dinv,
                         const float* __restrict__ bias, float* __restrict__ Agg,
                         int Nn, int C) {
  int i = blockIdx.x * blockDim.x + threadIdx.x;
  int total = Nn * C;
  if (i >= total) return;
  int r = i / C;
  int c = i - r * C;
  float ns = dinv[r] * dinv[r];
  Agg[i] = Ht[i] * ns + bias[c];
}

// ---------------------------------------------------------------------------
// Edge scatter-add: Agg[dst] += Ht[src] * dinv[src]*dinv[dst]
// Thread handles 4 channels of one edge (channel-fast for coalescing).
// ---------------------------------------------------------------------------
__global__ void edge_agg(const float* __restrict__ Ht, const int* __restrict__ src,
                         const int* __restrict__ dst, const float* __restrict__ dinv,
                         float* __restrict__ Agg, int E, int C) {
  int per = C >> 2;                 // float4 chunks per edge
  int total = E * per;
  int i = blockIdx.x * blockDim.x + threadIdx.x;
  if (i >= total) return;
  int e = i / per;
  int c = (i - e * per) << 2;
  int s = src[e], d = dst[e];
  float wgt = dinv[s] * dinv[d];
  const float4 v = *reinterpret_cast<const float4*>(&Ht[(size_t)s * C + c]);
  float* p = &Agg[(size_t)d * C + c];
  unsafeAtomicAdd(p + 0, v.x * wgt);
  unsafeAtomicAdd(p + 1, v.y * wgt);
  unsafeAtomicAdd(p + 2, v.z * wgt);
  unsafeAtomicAdd(p + 3, v.w * wgt);
}

// ---------------------------------------------------------------------------
// BatchNorm statistics: per-channel sum / sumsq. blockDim.x == C (256).
// ---------------------------------------------------------------------------
#define BN_ROWS 512
__global__ void bn_stats(const float* __restrict__ Agg, float* __restrict__ sum,
                         float* __restrict__ sumsq, int Nn, int C) {
  int c  = threadIdx.x;
  int r0 = blockIdx.x * BN_ROWS;
  int r1 = r0 + BN_ROWS; if (r1 > Nn) r1 = Nn;
  float s = 0.0f, q = 0.0f;
  for (int r = r0; r < r1; ++r) {
    float v = Agg[(size_t)r * C + c];
    s += v; q += v * v;
  }
  unsafeAtomicAdd(&sum[c], s);
  unsafeAtomicAdd(&sumsq[c], q);
}

__global__ void bn_finalize(const float* __restrict__ sum, const float* __restrict__ sumsq,
                            const float* __restrict__ gamma, const float* __restrict__ beta,
                            float* __restrict__ scale, float* __restrict__ shift, int Nn) {
  int c = threadIdx.x;
  float inv_n = 1.0f / (float)Nn;
  float mu  = sum[c] * inv_n;
  float var = sumsq[c] * inv_n - mu * mu;           // biased variance
  float sc  = gamma[c] * rsqrtf(var + 1e-5f);
  scale[c] = sc;
  shift[c] = beta[c] - mu * sc;
}

// ---------------------------------------------------------------------------
// Fused BN-apply + ReLU + train-mode dropout (p=0.5, scale 2x, hash RNG)
// ---------------------------------------------------------------------------
__global__ void bn_relu_drop(const float* __restrict__ Agg, const float* __restrict__ scale,
                             const float* __restrict__ shift, float* __restrict__ H,
                             int Nn, int C, unsigned layer) {
  int i = blockIdx.x * blockDim.x + threadIdx.x;
  int total = Nn * C;
  if (i >= total) return;
  int r = i / C;
  int c = i - r * C;
  float v = Agg[i] * scale[c] + shift[c];
  v = v > 0.0f ? v : 0.0f;
  unsigned h = mix32((unsigned)i ^ (layer * 0x9E3779B9u + 0x85EBCA6Bu));
  H[i] = (h & 1u) ? (v * 2.0f) : 0.0f;
}

// ---------------------------------------------------------------------------
// Row-wise log-softmax (C <= 64). One wave32 per row; 8 rows per block.
// ---------------------------------------------------------------------------
__global__ void logsoftmax_rows(const float* __restrict__ Agg, float* __restrict__ out,
                                int Nn, int C) {
  int lane = threadIdx.x & 31;
  int wv   = threadIdx.x >> 5;
  int row  = blockIdx.x * 8 + wv;
  if (row >= Nn) return;
  const float* p = Agg + (size_t)row * C;
  float v0 = (lane      < C) ? p[lane]      : -FLT_MAX;
  float v1 = (lane + 32 < C) ? p[lane + 32] : -FLT_MAX;
  float m = fmaxf(v0, v1);
#pragma unroll
  for (int off = 16; off > 0; off >>= 1) m = fmaxf(m, __shfl_xor(m, off, 32));
  float e = ((lane < C) ? __expf(v0 - m) : 0.0f) +
            ((lane + 32 < C) ? __expf(v1 - m) : 0.0f);
#pragma unroll
  for (int off = 16; off > 0; off >>= 1) e += __shfl_xor(e, off, 32);
  float ls = m + __logf(e);
  float* q = out + (size_t)row * C;
  if (lane < C)      q[lane]      = v0 - ls;
  if (lane + 32 < C) q[lane + 32] = v1 - ls;
}

// ---------------------------------------------------------------------------
// Host launcher
// ---------------------------------------------------------------------------
extern "C" void kernel_launch(void* const* d_in, const int* in_sizes, int n_in,
                              void* d_out, int out_size, void* d_ws, size_t ws_size,
                              hipStream_t stream) {
  (void)n_in; (void)out_size; (void)ws_size;
  const float* x     = (const float*)d_in[0];
  const int*   esrc  = (const int*)d_in[1];
  const int*   edst  = (const int*)d_in[2];
  const float* W0    = (const float*)d_in[3];
  const float* b0    = (const float*)d_in[4];
  const float* W1    = (const float*)d_in[5];
  const float* b1    = (const float*)d_in[6];
  const float* W4    = (const float*)d_in[7];
  const float* b4    = (const float*)d_in[8];
  const float* gamma = (const float*)d_in[9];
  const float* beta  = (const float*)d_in[10];

  const int E    = in_sizes[1];
  const int HID  = in_sizes[4];              // 256
  const int IN   = in_sizes[3] / HID;        // 128
  const int OUTC = in_sizes[8];              // 40
  const int Nn   = in_sizes[0] / IN;         // 100000

  // Workspace layout (floats)
  float* ws = (float*)d_ws;
  size_t o = 0;
  float* Ht   = ws + o; o += (size_t)Nn * HID;
  float* Agg  = ws + o; o += (size_t)Nn * HID;
  float* H    = ws + o; o += (size_t)Nn * HID;
  float* dinv = ws + o; o += (size_t)((Nn + 255) & ~255);
  float* bnsum = ws + o; o += HID;
  float* bnsq  = ws + o; o += HID;
  float* scl   = ws + o; o += HID;
  float* shf   = ws + o; o += HID;

  const int TPB = 256;
  // ---- degrees / normalization ----
  zero_f32<<<(Nn + TPB - 1) / TPB, TPB, 0, stream>>>(dinv, Nn);
  degree_kernel<<<(E + TPB - 1) / TPB, TPB, 0, stream>>>(edst, dinv, E);
  dinv_kernel<<<(Nn + TPB - 1) / TPB, TPB, 0, stream>>>(dinv, Nn);

  const float* hin = x;
  int K = IN;
  for (int l = 0; l < 4; ++l) {
    const float* W = (l == 0) ? W0 : W1;
    const float* b = (l == 0) ? b0 : b1;

    dim3 ggrid((Nn + MBLK - 1) / MBLK, (HID + NBLK - 1) / NBLK);
    gemm_wmma_bf16<<<ggrid, 256, 0, stream>>>(hin, W, Ht, Nn, K, HID);

    int totNC = Nn * HID;
    init_agg<<<(totNC + TPB - 1) / TPB, TPB, 0, stream>>>(Ht, dinv, b, Agg, Nn, HID);

    int totE = E * (HID >> 2);
    edge_agg<<<(totE + TPB - 1) / TPB, TPB, 0, stream>>>(Ht, esrc, edst, dinv, Agg, E, HID);

    zero_f32<<<(2 * HID + TPB - 1) / TPB, TPB, 0, stream>>>(bnsum, 2 * HID); // bnsum+bnsq contiguous
    bn_stats<<<(Nn + BN_ROWS - 1) / BN_ROWS, HID, 0, stream>>>(Agg, bnsum, bnsq, Nn, HID);
    bn_finalize<<<1, HID, 0, stream>>>(bnsum, bnsq, gamma, beta, scl, shf, Nn);
    bn_relu_drop<<<(totNC + TPB - 1) / TPB, TPB, 0, stream>>>(Agg, scl, shf, H, Nn, HID, (unsigned)l);

    hin = H;
    K = HID;
  }

  // ---- final conv (HID -> OUTC) + log-softmax ----
  {
    dim3 ggrid((Nn + MBLK - 1) / MBLK, (OUTC + NBLK - 1) / NBLK);
    gemm_wmma_bf16<<<ggrid, 256, 0, stream>>>(H, W4, Ht, Nn, HID, OUTC);

    int totNC = Nn * OUTC;
    init_agg<<<(totNC + TPB - 1) / TPB, TPB, 0, stream>>>(Ht, dinv, b4, Agg, Nn, OUTC);

    int totE = E * (OUTC >> 2);
    edge_agg<<<(totE + TPB - 1) / TPB, TPB, 0, stream>>>(Ht, esrc, edst, dinv, Agg, E, OUTC);

    logsoftmax_rows<<<(Nn + 7) / 8, 256, 0, stream>>>(Agg, (float*)d_out, Nn, OUTC);
  }
}
